// MambaOnlyVIX_35828617183487
// MI455X (gfx1250) — matrix operand order repaired and checked by reading, not compiled, not run on hardware
//
#include <hip/hip_runtime.h>
#include <math.h>

#define NFEAT 15
#define D_MODEL 256
#define N_LAYERS 4
#define D_STATE 64
#define D_CONV 4
#define D_INNER 512
#define DT_RANK 16
#define HID 128
#define NBUCK 15
#define BATCH 8
#define TLEN 2048
#define TOK (BATCH * TLEN)   // 16384 tokens

typedef __attribute__((ext_vector_type(16))) _Float16 v16h;
typedef __attribute__((ext_vector_type(8)))  _Float16 v8h;
typedef __attribute__((ext_vector_type(8)))  float    v8f;

// ---------------------------------------------------------------------------
// Block reduction helpers (wave32).  NW = number of waves in the block.
// Leading __syncthreads protects `red` reuse across back-to-back reductions.
// ---------------------------------------------------------------------------
template <int NW>
__device__ __forceinline__ float blk_sum(float v, float* red) {
#pragma unroll
  for (int off = 16; off; off >>= 1) v += __shfl_xor(v, off, 32);
  int w = threadIdx.x >> 5;
  __syncthreads();
  if ((threadIdx.x & 31) == 0) red[w] = v;
  __syncthreads();
  float s = 0.f;
#pragma unroll
  for (int k = 0; k < NW; ++k) s += red[k];
  return s;
}

template <int NW>
__device__ __forceinline__ float blk_max(float v, float* red) {
#pragma unroll
  for (int off = 16; off; off >>= 1) v = fmaxf(v, __shfl_xor(v, off, 32));
  int w = threadIdx.x >> 5;
  __syncthreads();
  if ((threadIdx.x & 31) == 0) red[w] = v;
  __syncthreads();
  float s = red[0];
#pragma unroll
  for (int k = 1; k < NW; ++k) s = fmaxf(s, red[k]);
  return s;
}

__device__ __forceinline__ float gelu_exact(float x) {
  return 0.5f * x * (1.f + erff(x * 0.70710678118654752f));
}
__device__ __forceinline__ float silu(float x) {
  return x / (1.f + __expf(-x));
}

// ---------------------------------------------------------------------------
// Bar projection: x = LN(gelu(LN(bars@W1^T+b1))@W2^T + b2) * mask
// One block (256 thr) per token.
// ---------------------------------------------------------------------------
__global__ __launch_bounds__(256) void k_barproj(
    const float* __restrict__ bars, const float* __restrict__ mask,
    const float* __restrict__ w1, const float* __restrict__ b1,
    const float* __restrict__ g1, const float* __restrict__ be1,
    const float* __restrict__ w2, const float* __restrict__ b2,
    const float* __restrict__ g2, const float* __restrict__ be2,
    float* __restrict__ X) {
  const int token = blockIdx.x;
  const int d = threadIdx.x;
  __shared__ float feat[NFEAT];
  __shared__ float row[D_MODEL];
  __shared__ float red[8];

  if (d < NFEAT) feat[d] = bars[(size_t)token * NFEAT + d];
  __syncthreads();

  float v = b1[d];
#pragma unroll
  for (int f = 0; f < NFEAT; ++f) v += feat[f] * w1[d * NFEAT + f];

  float m = blk_sum<8>(v, red) * (1.f / D_MODEL);
  float var = blk_sum<8>((v - m) * (v - m), red) * (1.f / D_MODEL);
  v = (v - m) * rsqrtf(var + 1e-5f) * g1[d] + be1[d];
  v = gelu_exact(v);

  row[d] = v;
  __syncthreads();

  float v2 = b2[d];
  const float* w2r = w2 + (size_t)d * D_MODEL;
  for (int j = 0; j < D_MODEL; ++j) v2 += row[j] * w2r[j];

  float m2 = blk_sum<8>(v2, red) * (1.f / D_MODEL);
  float var2 = blk_sum<8>((v2 - m2) * (v2 - m2), red) * (1.f / D_MODEL);
  v2 = (v2 - m2) * rsqrtf(var2 + 1e-5f) * g2[d] + be2[d];

  X[(size_t)token * D_MODEL + d] = v2 * mask[token];
}

// ---------------------------------------------------------------------------
// f32 -> f16 conversion (4 elems/thread, vectorized read)
// ---------------------------------------------------------------------------
__global__ void k_cvt(const float* __restrict__ s, _Float16* __restrict__ d, int n) {
  int i = (blockIdx.x * 256 + threadIdx.x) * 4;
  if (i + 3 < n) {
    float4 f = *(const float4*)(s + i);
    d[i + 0] = (_Float16)f.x;
    d[i + 1] = (_Float16)f.y;
    d[i + 2] = (_Float16)f.z;
    d[i + 3] = (_Float16)f.w;
  }
}

// ---------------------------------------------------------------------------
// WMMA GEMM:  C[M,N] = A[M,K] @ W[N,K]^T   (f16 operands, f32 accumulate)
//
// Block = 256 threads = 8 waves.  The whole block shares one 64-column W
// panel, staged in LDS in K-chunks of 256 (64 x 264 f16 = 33 KB, 16B row
// skew for bank spread).  Each wave owns a 32x64 output tile: 2 A fragments
// x 4 B fragments -> 8 v_wmma_f32_16x16x32_f16 per K-step.  A streams from
// global (rows are wave-private), B comes from ds_load_b128.
// Fragment layouts per CDNA5 ISA §7.12.2.
// ---------------------------------------------------------------------------
#define KC 256           // K-chunk staged in LDS (K is 256 or 512 here)
#define BPITCH (KC + 8)  // halves per LDS row (16B skew)

__global__ __launch_bounds__(256) void k_gemm_wmma(
    const _Float16* __restrict__ A, const _Float16* __restrict__ W,
    float* __restrict__ C, int M, int N, int K) {
  __shared__ _Float16 Bsh[64 * BPITCH];  // 33,792 B

  const int tid = threadIdx.x;
  const int lane = tid & 31;
  const int wave = tid >> 5;
  const int i0 = (blockIdx.x * 8 + wave) * 32;
  const int j0 = blockIdx.y * 64;

  const int row = lane & 15;     // A row / B col / D col within 16x16 tile
  const int khalf = lane >> 4;   // which K-half this lane owns

  v8f acc[2][4] = {};

  // A fragment base pointers: rows i0+row and i0+16+row
  const _Float16* Ap[2];
  Ap[0] = A + (size_t)(i0 + row) * K + khalf * 8;
  Ap[1] = A + (size_t)(i0 + 16 + row) * K + khalf * 8;

  // B fragment base pointers in LDS (col-local n = c*16+row)
  const _Float16* Bp[4];
#pragma unroll
  for (int c = 0; c < 4; ++c)
    Bp[c] = Bsh + (size_t)(c * 16 + row) * BPITCH + khalf * 8;

  // cooperative-load mapping: 4 threads per W row, 64 halves (128B) each
  const int lrow = tid >> 2;
  const int seg = tid & 3;
  const int gn = j0 + lrow;
  const _Float16* wsrc = W + (size_t)((gn < N) ? gn : 0) * K + seg * 64;
  _Float16* wdst = Bsh + (size_t)lrow * BPITCH + seg * 64;

  for (int kc = 0; kc < K; kc += KC) {
    __syncthreads();  // previous chunk fully consumed
#pragma unroll
    for (int j = 0; j < 8; ++j)
      *(v8h*)(wdst + j * 8) = *(const v8h*)(wsrc + kc + j * 8);
    __syncthreads();  // chunk staged

#pragma unroll 4
    for (int kk = 0; kk < KC; kk += 32) {
      const int k0 = kc + kk;
      if (k0 + 64 < K) {
        __builtin_prefetch(Ap[0] + k0 + 64, 0, 1);  // global_prefetch_b8
        __builtin_prefetch(Ap[1] + k0 + 64, 0, 1);
      }

      v16h a[2];
#pragma unroll
      for (int rr = 0; rr < 2; ++rr) {
        v8h lo = *(const v8h*)(Ap[rr] + k0);
        v8h hi = *(const v8h*)(Ap[rr] + k0 + 16);
#pragma unroll
        for (int e = 0; e < 8; ++e) { a[rr][e] = lo[e]; a[rr][e + 8] = hi[e]; }
      }

      v16h b[4];
#pragma unroll
      for (int c = 0; c < 4; ++c) {
        v8h lo = *(const v8h*)(Bp[c] + kk);
        v8h hi = *(const v8h*)(Bp[c] + kk + 16);
#pragma unroll
        for (int e = 0; e < 8; ++e) { b[c][e] = lo[e]; b[c][e + 8] = hi[e]; }
      }

#pragma unroll
      for (int rr = 0; rr < 2; ++rr)
#pragma unroll
        for (int c = 0; c < 4; ++c)
          acc[rr][c] = __builtin_amdgcn_wmma_f32_16x16x32_f16(
              false, a[rr], false, b[c], (short)0, acc[rr][c], false, false);
    }
  }

  // D layout: VGPR r -> row (khalf*8 + r), col = lane&15 within a 16x16 tile
  const int rbase = khalf * 8;
#pragma unroll
  for (int rr = 0; rr < 2; ++rr) {
#pragma unroll
    for (int c = 0; c < 4; ++c) {
      int n = j0 + c * 16 + row;
      if (n < N) {
#pragma unroll
        for (int r = 0; r < 8; ++r)
          C[(size_t)(i0 + rr * 16 + rbase + r) * N + n] = acc[rr][c][r];
      }
    }
  }
}

// ---------------------------------------------------------------------------
// Depthwise causal conv (D_CONV=4) over time + silu.
// XZ holds [x|z]; we read x half.  One thread per (token, channel).
// ---------------------------------------------------------------------------
__global__ void k_conv(const float* __restrict__ XZ, const float* __restrict__ cw,
                       const float* __restrict__ cb, float* __restrict__ XC) {
  int idx = blockIdx.x * 256 + threadIdx.x;
  int c = idx & (D_INNER - 1);
  int token = idx >> 9;
  int t = token & (TLEN - 1);
  float acc = cb[c];
#pragma unroll
  for (int k = 0; k < D_CONV; ++k) {
    int tt = t - (D_CONV - 1) + k;
    if (tt >= 0)
      acc += cw[c * D_CONV + k] * XZ[(size_t)(token - (D_CONV - 1) + k) * (2 * D_INNER) + c];
  }
  XC[(size_t)token * D_INNER + c] = silu(acc);
}

// ---------------------------------------------------------------------------
// dt = softplus(dbl[:, :16] @ dt_w^T + dt_b).  One thread per (token, channel).
// ---------------------------------------------------------------------------
__global__ void k_dt(const float* __restrict__ DBL, const float* __restrict__ dtw,
                     const float* __restrict__ dtb, float* __restrict__ DT) {
  int idx = blockIdx.x * 256 + threadIdx.x;
  int c = idx & (D_INNER - 1);
  int token = idx >> 9;
  const float* db = DBL + (size_t)token * (DT_RANK + 2 * D_STATE);
  float acc = dtb[c];
#pragma unroll
  for (int r = 0; r < DT_RANK; ++r) acc += db[r] * dtw[c * DT_RANK + r];
  float sp = (acc > 20.f) ? acc : log1pf(__expf(acc));
  DT[(size_t)token * D_INNER + c] = sp;
}

// ---------------------------------------------------------------------------
// Selective scan.  Block = 1024 thr = 32 waves.  Wave handles one channel d;
// lane owns 2 of the 64 states; wave-level shfl_xor reduction per step.
// grid = (D_INNER/32, BATCH).  Latency-critical serial loop, no barriers.
// ---------------------------------------------------------------------------
__global__ __launch_bounds__(1024) void k_scan(
    const float* __restrict__ DT, const float* __restrict__ XC,
    const float* __restrict__ DBL, const float* __restrict__ Alog,
    float* __restrict__ Y) {
  const int b = blockIdx.y;
  const int d = blockIdx.x * 32 + (threadIdx.x >> 5);
  const int lane = threadIdx.x & 31;
  const int s0 = lane, s1 = lane + 32;

  const float A0 = -__expf(Alog[(size_t)d * D_STATE + s0]);
  const float A1 = -__expf(Alog[(size_t)d * D_STATE + s1]);
  float h0 = 0.f, h1 = 0.f;
  const size_t tokbase = (size_t)b * TLEN;

  for (int t = 0; t < TLEN; ++t) {
    const size_t token = tokbase + t;
    const float dt = DT[token * D_INNER + d];
    const float x = XC[token * D_INNER + d];
    const float* dbl = DBL + token * (DT_RANK + 2 * D_STATE);
    const float B0 = dbl[DT_RANK + s0], B1 = dbl[DT_RANK + s1];
    const float C0 = dbl[DT_RANK + D_STATE + s0], C1 = dbl[DT_RANK + D_STATE + s1];
    const float dx = dt * x;
    h0 = __expf(dt * A0) * h0 + dx * B0;
    h1 = __expf(dt * A1) * h1 + dx * B1;
    float p = h0 * C0 + h1 * C1;
#pragma unroll
    for (int off = 16; off; off >>= 1) p += __shfl_xor(p, off, 32);
    if (lane == 0) Y[token * D_INNER + d] = p;
  }
}

// ---------------------------------------------------------------------------
// Gate: Y = (Y + XC*Dp) * silu(z)  (in place)
// ---------------------------------------------------------------------------
__global__ void k_gate(float* __restrict__ Y, const float* __restrict__ XC,
                       const float* __restrict__ XZ, const float* __restrict__ Dp) {
  int idx = blockIdx.x * 256 + threadIdx.x;
  int c = idx & (D_INNER - 1);
  int token = idx >> 9;
  size_t i = (size_t)token * D_INNER + c;
  float z = XZ[(size_t)token * (2 * D_INNER) + D_INNER + c];
  Y[i] = (Y[i] + XC[i] * Dp[c]) * silu(z);
}

// ---------------------------------------------------------------------------
// LayerNorm + residual: X = LN(OUT)*g + b + X.  Block per token.
// ---------------------------------------------------------------------------
__global__ __launch_bounds__(256) void k_lnres(
    const float* __restrict__ OUT, const float* __restrict__ g,
    const float* __restrict__ be, float* __restrict__ X) {
  const int token = blockIdx.x;
  const int d = threadIdx.x;
  __shared__ float red[8];
  float v = OUT[(size_t)token * D_MODEL + d];
  float m = blk_sum<8>(v, red) * (1.f / D_MODEL);
  float var = blk_sum<8>((v - m) * (v - m), red) * (1.f / D_MODEL);
  float nv = (v - m) * rsqrtf(var + 1e-5f) * g[d] + be[d];
  size_t i = (size_t)token * D_MODEL + d;
  X[i] = nv + X[i];
}

// ---------------------------------------------------------------------------
// Attention pooling: logits over T, softmax, weighted sum -> hp[b, 256].
// One block (256 thr) per batch element.
// ---------------------------------------------------------------------------
__global__ __launch_bounds__(256) void k_pool(
    const float* __restrict__ X, const float* __restrict__ pw,
    const float* __restrict__ pb, float* __restrict__ HP) {
  const int b = blockIdx.x;
  __shared__ float l[TLEN];
  __shared__ float pwS[D_MODEL];
  __shared__ float red[8];
  const float* Xb = X + (size_t)b * TLEN * D_MODEL;

  pwS[threadIdx.x] = pw[threadIdx.x];
  __syncthreads();

  const float pbias = pb[0];
  for (int t = threadIdx.x; t < TLEN; t += 256) {
    float acc = pbias;
    const float* xr = Xb + (size_t)t * D_MODEL;
    for (int d = 0; d < D_MODEL; ++d) acc += xr[d] * pwS[d];
    l[t] = acc;
  }
  __syncthreads();

  float lm = -1e30f;
  for (int t = threadIdx.x; t < TLEN; t += 256) lm = fmaxf(lm, l[t]);
  float mx = blk_max<8>(lm, red);

  float ls = 0.f;
  for (int t = threadIdx.x; t < TLEN; t += 256) {
    float e = __expf(l[t] - mx);
    l[t] = e;
    ls += e;
  }
  float ssum = blk_sum<8>(ls, red);
  float inv = 1.f / ssum;
  __syncthreads();

  const int d = threadIdx.x;
  float acc = 0.f;
  for (int t = 0; t < TLEN; ++t) acc += l[t] * Xb[(size_t)t * D_MODEL + d];
  HP[(size_t)b * D_MODEL + d] = acc * inv;
}

// ---------------------------------------------------------------------------
// Heads: vix regression + bucket classification.  Block (128 thr) per batch.
// d_out layout: [0..7] = vix_pred, [8..127] = bucket_logits (8x15).
// ---------------------------------------------------------------------------
__global__ __launch_bounds__(128) void k_head(
    const float* __restrict__ HP,
    const float* __restrict__ hw1, const float* __restrict__ hb1,
    const float* __restrict__ hg1, const float* __restrict__ hbe1,
    const float* __restrict__ hw2, const float* __restrict__ hb2,
    const float* __restrict__ hg2, const float* __restrict__ hbe2,
    const float* __restrict__ hw3, const float* __restrict__ hb3,
    const float* __restrict__ cw1, const float* __restrict__ cb1,
    const float* __restrict__ cg1, const float* __restrict__ cbe1,
    const float* __restrict__ cw2, const float* __restrict__ cb2,
    float* __restrict__ out) {
  const int b = blockIdx.x;
  const int i = threadIdx.x;
  __shared__ float hp[D_MODEL];
  __shared__ float h1s[HID], h2s[HID], c1s[HID];
  __shared__ float red[4];

  hp[i] = HP[(size_t)b * D_MODEL + i];
  hp[i + HID] = HP[(size_t)b * D_MODEL + i + HID];
  __syncthreads();

  // h1 = gelu(LN(hp @ hw1^T + hb1))
  float v = hb1[i];
  for (int d = 0; d < D_MODEL; ++d) v += hp[d] * hw1[i * D_MODEL + d];
  float m = blk_sum<4>(v, red) * (1.f / HID);
  float var = blk_sum<4>((v - m) * (v - m), red) * (1.f / HID);
  v = gelu_exact((v - m) * rsqrtf(var + 1e-5f) * hg1[i] + hbe1[i]);
  h1s[i] = v;
  __syncthreads();

  // h2 = gelu(LN(h1 @ hw2^T + hb2))
  float v2 = hb2[i];
  for (int j = 0; j < HID; ++j) v2 += h1s[j] * hw2[i * HID + j];
  float m2 = blk_sum<4>(v2, red) * (1.f / HID);
  float var2 = blk_sum<4>((v2 - m2) * (v2 - m2), red) * (1.f / HID);
  v2 = gelu_exact((v2 - m2) * rsqrtf(var2 + 1e-5f) * hg2[i] + hbe2[i]);
  h2s[i] = v2;
  __syncthreads();

  float p = h2s[i] * hw3[i];
  float vix = blk_sum<4>(p, red);
  if (i == 0) out[b] = vix + hb3[0];

  // c1 = gelu(LN(hp @ cw1^T + cb1))
  float cv = cb1[i];
  for (int d = 0; d < D_MODEL; ++d) cv += hp[d] * cw1[i * D_MODEL + d];
  float cm = blk_sum<4>(cv, red) * (1.f / HID);
  float cvar = blk_sum<4>((cv - cm) * (cv - cm), red) * (1.f / HID);
  cv = gelu_exact((cv - cm) * rsqrtf(cvar + 1e-5f) * cg1[i] + cbe1[i]);
  c1s[i] = cv;
  __syncthreads();

  if (i < NBUCK) {
    float o = cb2[i];
    for (int j = 0; j < HID; ++j) o += c1s[j] * cw2[i * HID + j];
    out[BATCH + b * NBUCK + i] = o;
  }
}

// ---------------------------------------------------------------------------
// Launch
// ---------------------------------------------------------------------------
extern "C" void kernel_launch(void* const* d_in, const int* in_sizes, int n_in,
                              void* d_out, int out_size, void* d_ws, size_t ws_size,
                              hipStream_t stream) {
  (void)in_sizes; (void)n_in; (void)out_size; (void)ws_size;

  const float* bars    = (const float*)d_in[0];
  const float* mask    = (const float*)d_in[1];
  const float* bp_w1   = (const float*)d_in[2];
  const float* bp_b1   = (const float*)d_in[3];
  const float* bp_g1   = (const float*)d_in[4];
  const float* bp_be1  = (const float*)d_in[5];
  const float* bp_w2   = (const float*)d_in[6];
  const float* bp_b2   = (const float*)d_in[7];
  const float* bp_g2   = (const float*)d_in[8];
  const float* bp_be2  = (const float*)d_in[9];
  const float* m_in_w  = (const float*)d_in[10];
  const float* m_conv_w= (const float*)d_in[11];
  const float* m_conv_b= (const float*)d_in[12];
  const float* m_xp_w  = (const float*)d_in[13];
  const float* m_dt_w  = (const float*)d_in[14];
  const float* m_dt_b  = (const float*)d_in[15];
  const float* m_Alog  = (const float*)d_in[16];
  const float* m_D     = (const float*)d_in[17];
  const float* m_out_w = (const float*)d_in[18];
  const float* m_ln_g  = (const float*)d_in[19];
  const float* m_ln_b  = (const float*)d_in[20];
  const float* pool_w  = (const float*)d_in[21];
  const float* pool_b  = (const float*)d_in[22];
  const float* h_w1    = (const float*)d_in[23];
  const float* h_b1    = (const float*)d_in[24];
  const float* h_g1    = (const float*)d_in[25];
  const float* h_be1   = (const float*)d_in[26];
  const float* h_w2    = (const float*)d_in[27];
  const float* h_b2    = (const float*)d_in[28];
  const float* h_g2    = (const float*)d_in[29];
  const float* h_be2   = (const float*)d_in[30];
  const float* h_w3    = (const float*)d_in[31];
  const float* h_b3    = (const float*)d_in[32];
  const float* c_w1    = (const float*)d_in[33];
  const float* c_b1    = (const float*)d_in[34];
  const float* c_g1    = (const float*)d_in[35];
  const float* c_be1   = (const float*)d_in[36];
  const float* c_w2    = (const float*)d_in[37];
  const float* c_b2    = (const float*)d_in[38];

  char* ws = (char*)d_ws;
  size_t off = 0;
  auto alloc = [&](size_t bytes) -> void* {
    void* p = ws + off;
    off += (bytes + 255) & ~(size_t)255;
    return p;
  };
  float*    X    = (float*)alloc((size_t)TOK * D_MODEL * 4);
  float*    XZ   = (float*)alloc((size_t)TOK * 2 * D_INNER * 4);
  float*    XC   = (float*)alloc((size_t)TOK * D_INNER * 4);
  float*    DBL  = (float*)alloc((size_t)TOK * (DT_RANK + 2 * D_STATE) * 4);
  float*    DT   = (float*)alloc((size_t)TOK * D_INNER * 4);
  float*    Y    = (float*)alloc((size_t)TOK * D_INNER * 4);
  float*    OUTb = (float*)alloc((size_t)TOK * D_MODEL * 4);
  _Float16* H16  = (_Float16*)alloc((size_t)TOK * D_INNER * 2);
  _Float16* W16  = (_Float16*)alloc((size_t)(2 * D_INNER) * D_MODEL * 2);
  float*    HP   = (float*)alloc((size_t)BATCH * D_MODEL * 4);

  auto cvt_grid = [](int n) { return (n / 4 + 255) / 256; };

  // Bar projection -> X
  k_barproj<<<TOK, 256, 0, stream>>>(bars, mask, bp_w1, bp_b1, bp_g1, bp_be1,
                                     bp_w2, bp_b2, bp_g2, bp_be2, X);

  for (int i = 0; i < N_LAYERS; ++i) {
    const float* in_w  = m_in_w  + (size_t)i * 2 * D_INNER * D_MODEL;
    const float* cw    = m_conv_w+ (size_t)i * D_INNER * D_CONV;
    const float* cb    = m_conv_b+ (size_t)i * D_INNER;
    const float* xpw   = m_xp_w  + (size_t)i * (DT_RANK + 2 * D_STATE) * D_INNER;
    const float* dtw   = m_dt_w  + (size_t)i * D_INNER * DT_RANK;
    const float* dtb   = m_dt_b  + (size_t)i * D_INNER;
    const float* alog  = m_Alog  + (size_t)i * D_INNER * D_STATE;
    const float* dp    = m_D     + (size_t)i * D_INNER;
    const float* ow    = m_out_w + (size_t)i * D_MODEL * D_INNER;
    const float* lng   = m_ln_g  + (size_t)i * D_MODEL;
    const float* lnb   = m_ln_b  + (size_t)i * D_MODEL;

    // in_proj: XZ = X @ in_w^T   (16384 x 1024, K=256)  [WMMA]
    k_cvt<<<cvt_grid(TOK * D_MODEL), 256, 0, stream>>>(X, H16, TOK * D_MODEL);
    k_cvt<<<cvt_grid(2 * D_INNER * D_MODEL), 256, 0, stream>>>(in_w, W16, 2 * D_INNER * D_MODEL);
    k_gemm_wmma<<<dim3(TOK / 256, (2 * D_INNER) / 64), 256, 0, stream>>>(
        H16, W16, XZ, TOK, 2 * D_INNER, D_MODEL);

    // depthwise conv + silu -> XC
    k_conv<<<TOK * D_INNER / 256, 256, 0, stream>>>(XZ, cw, cb, XC);

    // x_proj: DBL = XC @ xpw^T   (16384 x 144, K=512)  [WMMA, ragged N]
    k_cvt<<<cvt_grid(TOK * D_INNER), 256, 0, stream>>>(XC, H16, TOK * D_INNER);
    k_cvt<<<cvt_grid((DT_RANK + 2 * D_STATE) * D_INNER), 256, 0, stream>>>(
        xpw, W16, (DT_RANK + 2 * D_STATE) * D_INNER);
    k_gemm_wmma<<<dim3(TOK / 256, (DT_RANK + 2 * D_STATE + 63) / 64), 256, 0, stream>>>(
        H16, W16, DBL, TOK, DT_RANK + 2 * D_STATE, D_INNER);

    // dt projection + softplus (K=16, scalar)
    k_dt<<<TOK * D_INNER / 256, 256, 0, stream>>>(DBL, dtw, dtb, DT);

    // selective scan (serial over T, 4096 waves)
    k_scan<<<dim3(D_INNER / 32, BATCH), 1024, 0, stream>>>(DT, XC, DBL, alog, Y);

    // gate: Y = (Y + XC*D) * silu(z)
    k_gate<<<TOK * D_INNER / 256, 256, 0, stream>>>(Y, XC, XZ, dp);

    // out_proj: OUT = Y @ ow^T   (16384 x 256, K=512)  [WMMA]
    k_cvt<<<cvt_grid(TOK * D_INNER), 256, 0, stream>>>(Y, H16, TOK * D_INNER);
    k_cvt<<<cvt_grid(D_MODEL * D_INNER), 256, 0, stream>>>(ow, W16, D_MODEL * D_INNER);
    k_gemm_wmma<<<dim3(TOK / 256, D_MODEL / 64), 256, 0, stream>>>(
        H16, W16, OUTb, TOK, D_MODEL, D_INNER);

    // LN + residual -> X
    k_lnres<<<TOK, 256, 0, stream>>>(OUTb, lng, lnb, X);
  }

  // pooling + heads
  k_pool<<<BATCH, 256, 0, stream>>>(X, pool_w, pool_b, HP);
  k_head<<<BATCH, 128, 0, stream>>>(HP, h_w1, h_b1, h_g1, h_be1, h_w2, h_b2,
                                    h_g2, h_be2, h_w3, h_b3, c_w1, c_b1, c_g1,
                                    c_be1, c_w2, c_b2, (float*)d_out);
}
